// ScaledDotProductAttention_27023934227039
// MI455X (gfx1250) — matrix operand order repaired and checked by reading, not compiled, not run on hardware
//
#include <hip/hip_runtime.h>
#include <hip/hip_bf16.h>

typedef __attribute__((ext_vector_type(16))) _Float16 v16h;
typedef __attribute__((ext_vector_type(8)))  _Float16 v8h;
typedef __attribute__((ext_vector_type(8)))  float    v8f;
typedef __attribute__((ext_vector_type(4)))  float    v4f;

#define B_   2
#define H_   16
#define S_   2048
#define D_   128
#define SCOL 2052            // 4-float pad: 16B-aligned rows, no LDS bank conflicts
#define NEG  (-1000000000.0f)
#define EPS  1e-12f

// ---------------------------------------------------------------------------
// Pass 1a: L2-normalize K across the HEAD axis (dim=1), emit f16.
// ---------------------------------------------------------------------------
__global__ void knorm_kernel(const float* __restrict__ k, _Float16* __restrict__ k16) {
    int gid = blockIdx.x * blockDim.x + threadIdx.x;      // over B*S*D
    if (gid >= B_ * S_ * D_) return;
    int d = gid % D_;
    int s = (gid / D_) % S_;
    int b = gid / (S_ * D_);
    float kv[H_];
    float ss = 0.0f;
#pragma unroll
    for (int h = 0; h < H_; ++h) {
        float x = k[((size_t)(b * H_ + h) * S_ + s) * D_ + d];
        kv[h] = x;
        ss += x * x;
    }
    float inv = 1.0f / fmaxf(sqrtf(ss), EPS);
#pragma unroll
    for (int h = 0; h < H_; ++h)
        k16[((size_t)(b * H_ + h) * S_ + s) * D_ + d] = (_Float16)(kv[h] * inv);
}

// ---------------------------------------------------------------------------
// Pass 1b: V [BH][S][D] f32  ->  Vt [BH][D][S] f16 (tiled LDS transpose).
// ---------------------------------------------------------------------------
__global__ void vtrans_kernel(const float* __restrict__ v, _Float16* __restrict__ v16t) {
    __shared__ float tile[32][33];
    int bh = blockIdx.z;
    int s0 = blockIdx.x * 32, d0 = blockIdx.y * 32;
    int tx = threadIdx.x, ty = threadIdx.y;               // 32 x 8
    const float* vb = v + (size_t)bh * S_ * D_;
#pragma unroll
    for (int i = 0; i < 4; ++i)
        tile[ty + 8 * i][tx] = vb[(size_t)(s0 + ty + 8 * i) * D_ + d0 + tx];
    __syncthreads();
    _Float16* ob = v16t + (size_t)bh * D_ * S_;
#pragma unroll
    for (int i = 0; i < 4; ++i)
        ob[(size_t)(d0 + ty + 8 * i) * S_ + s0 + tx] = (_Float16)tile[tx][ty + 8 * i];
}

// ---------------------------------------------------------------------------
// Pass 2: attention. One workgroup (8 waves) per (b, h, 16-row q tile).
// ---------------------------------------------------------------------------
__global__ __launch_bounds__(256)
void attn_kernel(const float* __restrict__ q,
                 const _Float16* __restrict__ k16,
                 const _Float16* __restrict__ v16t,
                 const int* __restrict__ maskp,
                 float* __restrict__ out,
                 float* __restrict__ attn) {
    __shared__ float    scores[16 * SCOL];   // 131,328 B
    __shared__ _Float16 q16[16 * D_];        //   4,096 B

    const int tid = threadIdx.x;
    const int q0 = blockIdx.x * 16;
    const int h  = blockIdx.y;
    const int b  = blockIdx.z;
    const int bh = b * H_ + h;

    // ---- stage Q tile (16x128, contiguous) into LDS as f16; one-touch -> NT
    const float* qb = q + ((size_t)bh * S_ + q0) * D_;
    for (int i = tid; i < 16 * D_; i += 256)
        q16[i] = (_Float16)__builtin_nontemporal_load(qb + i);
    __syncthreads();

    const int wave = tid >> 5;
    const int lane = tid & 31;
    const int mrow = lane & 15;      // A-frag row / C-frag N column
    const int hi   = lane >> 4;      // lane-half selects K/M sub-block
    const int ncol = mrow;

    // ---- build 4 Q A-fragments (K = 0..127), contiguous k = 16*hi + s (matches B)
    v16h af[4];
    {
        const _Float16* qrow = q16 + mrow * D_;
#pragma unroll
        for (int f = 0; f < 4; ++f) {
            v8h lo = *(const v8h*)(qrow + f * 32 + hi * 16);
            v8h hh = *(const v8h*)(qrow + f * 32 + hi * 16 + 8);
            v16h a;
#pragma unroll
            for (int i = 0; i < 8; ++i) { a[i] = lo[i]; a[8 + i] = hh[i]; }
            af[f] = a;
        }
    }

    // ---- QK^T: each wave owns 16 key-tiles; double-buffered B-fragments
    const _Float16* kb = k16 + (size_t)bh * S_ * D_;
    const int* mb = maskp + b * S_;
    const int kt0 = wave * 16;

    // per-lane 16-bit mask bitmap: bit t <=> mask[key(kt0+t, ncol)] != 0.
    // (plain shift/and in the loop; avoids M0/v_movrels register-array indexing)
    unsigned mbits = 0;
#pragma unroll
    for (int t = 0; t < 16; ++t)
        mbits |= (mb[(kt0 + t) * 16 + ncol] != 0 ? 1u : 0u) << t;

    v16h bq[2][4];
    {
        const _Float16* krow = kb + (size_t)(kt0 * 16 + ncol) * D_;
#pragma unroll
        for (int f = 0; f < 4; ++f)
            bq[0][f] = *(const v16h*)(krow + f * 32 + hi * 16);
    }
#pragma unroll                                   // full unroll: constant buffer indices
    for (int t = 0; t < 16; ++t) {
        const int cur = t & 1, nxt = cur ^ 1;
        if (t + 1 < 16) {                        // prefetch next tile's B-frags
            const _Float16* krow = kb + (size_t)((kt0 + t + 1) * 16 + ncol) * D_;
#pragma unroll
            for (int f = 0; f < 4; ++f)
                bq[nxt][f] = *(const v16h*)(krow + f * 32 + hi * 16);
        }
        v8f acc = {};
#pragma unroll
        for (int f = 0; f < 4; ++f)
            acc = __builtin_amdgcn_wmma_f32_16x16x32_f16(
                false, af[f], false, bq[cur][f], (short)0, acc, false, false);
        const int key  = (kt0 + t) * 16 + ncol;
        const bool keep = ((mbits >> t) & 1u) != 0u;
#pragma unroll
        for (int j = 0; j < 8; ++j)              // D row m = j + hi*8
            scores[(j + hi * 8) * SCOL + key] = keep ? acc[j] : NEG;
    }
    __syncthreads();

    // ---- softmax per row: 16 threads/row, half-wave shfl reductions
    {
        const int r  = tid >> 4;
        const int c0 = tid & 15;
        float* srow = scores + r * SCOL;
        float mx = -3.4e38f;
        for (int c = c0; c < S_; c += 16) mx = fmaxf(mx, srow[c]);
#pragma unroll
        for (int mk = 1; mk < 16; mk <<= 1) mx = fmaxf(mx, __shfl_xor(mx, mk, 16));
        float sum = 0.0f;
        for (int c = c0; c < S_; c += 16) {
            float e = __expf(srow[c] - mx);
            srow[c] = e;
            sum += e;
        }
#pragma unroll
        for (int mk = 1; mk < 16; mk <<= 1) sum += __shfl_xor(sum, mk, 16);
        const float inv = 1.0f / sum;
        float* arow = attn + ((size_t)bh * S_ + q0 + r) * S_;
        for (int c = c0; c < S_; c += 16) {
            float p = srow[c] * inv;
            srow[c] = p;                                  // keep normalized P for PV
            __builtin_nontemporal_store(p, arow + c);     // 537MB stream: don't pollute L2
        }
    }
    __syncthreads();

    // ---- PV: each wave owns one 16-wide D-column tile; double-buffered
    {
        const int d0 = wave * 16;
        const _Float16* vb2 = v16t + ((size_t)bh * D_ + d0 + ncol) * S_;
        const float* prow = scores + mrow * SCOL;

        v16h bv[2];
        v4f pl0[2], pl1[2], ph0[2], ph1[2];
        {
            const float* pk = prow;
            pl0[0] = *(const v4f*)(pk + hi * 16);
            pl1[0] = *(const v4f*)(pk + hi * 16 + 4);
            ph0[0] = *(const v4f*)(pk + hi * 16 + 8);
            ph1[0] = *(const v4f*)(pk + hi * 16 + 12);
            bv[0]  = *(const v16h*)(vb2 + hi * 16);
        }
        v8f acc = {};
#pragma unroll 2                                 // ping-pong indices stay constant
        for (int kt2 = 0; kt2 < 64; ++kt2) {
            const int cur = kt2 & 1, nxt = cur ^ 1;
            if (kt2 + 1 < 64) {                  // prefetch next K-chunk
                const float* pk = prow + (kt2 + 1) * 32;
                pl0[nxt] = *(const v4f*)(pk + hi * 16);
                pl1[nxt] = *(const v4f*)(pk + hi * 16 + 4);
                ph0[nxt] = *(const v4f*)(pk + hi * 16 + 8);
                ph1[nxt] = *(const v4f*)(pk + hi * 16 + 12);
                bv[nxt]  = *(const v16h*)(vb2 + (kt2 + 1) * 32 + hi * 16);
            }
            v16h a;
#pragma unroll
            for (int i = 0; i < 4; ++i) {
                a[i]      = (_Float16)pl0[cur][i];
                a[4 + i]  = (_Float16)pl1[cur][i];
                a[8 + i]  = (_Float16)ph0[cur][i];
                a[12 + i] = (_Float16)ph1[cur][i];
            }
            acc = __builtin_amdgcn_wmma_f32_16x16x32_f16(
                false, a, false, bv[cur], (short)0, acc, false, false);
        }
        float* ob = out + ((size_t)bh * S_ + q0) * D_;
#pragma unroll
        for (int j = 0; j < 8; ++j)
            __builtin_nontemporal_store(acc[j], ob + (size_t)(j + hi * 8) * D_ + d0 + ncol);
    }
}

// ---------------------------------------------------------------------------
extern "C" void kernel_launch(void* const* d_in, const int* in_sizes, int n_in,
                              void* d_out, int out_size, void* d_ws, size_t ws_size,
                              hipStream_t stream) {
    const float* q    = (const float*)d_in[0];
    const float* k    = (const float*)d_in[1];
    const float* v    = (const float*)d_in[2];
    const int*   mask = (const int*)d_in[3];

    float* out  = (float*)d_out;                          // [B,H,S,D]
    float* attn = out + (size_t)B_ * H_ * S_ * D_;        // [B,H,S,S]

    // workspace: f16 normalized K, then f16 transposed V (33.5 MB total)
    _Float16* k16  = (_Float16*)d_ws;
    _Float16* v16t = k16 + (size_t)B_ * H_ * S_ * D_;

    knorm_kernel<<<(B_ * S_ * D_ + 255) / 256, 256, 0, stream>>>(k, k16);
    vtrans_kernel<<<dim3(S_ / 32, D_ / 32, B_ * H_), dim3(32, 8), 0, stream>>>(v, v16t);
    attn_kernel<<<dim3(S_ / 16, H_, B_), 256, 0, stream>>>(q, k16, v16t, mask, out, attn);
}
